// CrossAttention_5884105195652
// MI455X (gfx1250) — matrix operand order repaired and checked by reading, not compiled, not run on hardware
//
#include <hip/hip_runtime.h>

// ---------------------------------------------------------------------------
// CDNA5 (gfx1250) cross-attention, bf16 WMMA throughout, f32 accumulation.
// Weights are pre-transposed/converted to bf16 once so the GEMM k-loop is a
// pure vector-copy + v_wmma pipeline (no scalar LDS stores, no cvt in loop).
// ---------------------------------------------------------------------------

typedef __attribute__((ext_vector_type(16))) __bf16 v16bf;
typedef __attribute__((ext_vector_type(8)))  __bf16 v8bf;
typedef __attribute__((ext_vector_type(8)))  float  v8f;

__device__ __forceinline__ unsigned short f2bf(float f) {
  union { float f; unsigned u; } c; c.f = f;
  unsigned u = c.u;
  unsigned r = u + 0x7FFFu + ((u >> 16) & 1u);   // round-to-nearest-even
  return (unsigned short)(r >> 16);
}

__device__ __forceinline__ unsigned pack2(float a, float b) {
  return (unsigned)f2bf(a) | ((unsigned)f2bf(b) << 16);
}

// Build a 16-element bf16 fragment from two 16-byte chunks (LDS or global).
__device__ __forceinline__ v16bf frag_ld(const unsigned short* lo,
                                         const unsigned short* hi) {
  v8bf a = *(const v8bf*)lo;
  v8bf b = *(const v8bf*)hi;
  v16bf r;
#pragma unroll
  for (int i = 0; i < 8; ++i) { r[i] = a[i]; r[i + 8] = b[i]; }
  return r;
}

// ---------------------------------------------------------------------------
// One-time weight prep: Wt[k][n] = bf16(W[n][k]).  64x64 tiles via LDS.
// ---------------------------------------------------------------------------
__global__ __launch_bounds__(256) void transpose_w_bf16(
    const float* __restrict__ W, unsigned short* __restrict__ Wt, int N, int K)
{
  __shared__ float T[64][65];
  const int k0 = blockIdx.x * 64, n0 = blockIdx.y * 64;
  const int r  = threadIdx.x >> 4;         // 0..15
  const int c4 = (threadIdx.x & 15) * 4;   // 0..60
#pragma unroll
  for (int rr = 0; rr < 4; ++rr) {
    const int row = r + rr * 16;           // n within tile
    const float4 f = *(const float4*)(W + (size_t)(n0 + row) * K + k0 + c4);
    T[row][c4 + 0] = f.x; T[row][c4 + 1] = f.y;
    T[row][c4 + 2] = f.z; T[row][c4 + 3] = f.w;
  }
  __syncthreads();
#pragma unroll
  for (int rr = 0; rr < 4; ++rr) {
    const int krow = r + rr * 16;          // k within tile
    uint2 o;
    o.x = pack2(T[c4 + 0][krow], T[c4 + 1][krow]);
    o.y = pack2(T[c4 + 2][krow], T[c4 + 3][krow]);
    *(uint2*)(Wt + (size_t)(k0 + krow) * N + n0 + c4) = o;
  }
}

// ---------------------------------------------------------------------------
// Generic GEMM:  C[m][n] = (sum_k A[m][k] * Wt[k][n] + bias[n]) * alpha
//   A_BF16 : 0 = A fp32 (converted while staging), 1 = A bf16
//   OMODE  : 0 = f32 row-major [M][N]
//            1 = bf16 split  ((b*16+h)*rowsPer + i)*64 + d      (q / val)
//            2 = bf16 splitT ((b*16+h)*64 + d)*rowsPer + i      (k, pre-T)
// Block tile 128x128, 8 waves of 64x32, K staged 32 at a time.
// ---------------------------------------------------------------------------
template<int A_BF16, int OMODE>
__global__ __launch_bounds__(256) void gemm_wmma(
    const void* __restrict__ Av, const unsigned short* __restrict__ WtG,
    const float* __restrict__ bias, float alpha,
    void* __restrict__ Cout, int M, int N, int K, int rowsPer)
{
  constexpr int AS = 40;   // A-tile LDS row stride (ushorts), 80B
  constexpr int WS = 136;  // Wt  LDS row stride (ushorts), 272B
  __shared__ __align__(16) unsigned short Alds[128 * AS];
  __shared__ __align__(16) unsigned short Wlds[32 * WS];

  const int tid  = threadIdx.x;
  const int wid  = tid >> 5;
  const int lane = tid & 31;
  const int sel  = lane >> 4;
  const int l15  = lane & 15;

  const int m0 = blockIdx.y * 128;
  const int n0 = blockIdx.x * 128;
  const int wm = (wid >> 2) * 64;
  const int wn = (wid & 3) * 32;

  v8f acc[4][2];
#pragma unroll
  for (int i = 0; i < 4; ++i)
#pragma unroll
    for (int j = 0; j < 2; ++j)
#pragma unroll
      for (int r = 0; r < 8; ++r) acc[i][j][r] = 0.0f;

  const int srow = tid >> 1;         // A staging row (0..127)
  const int sks  = (tid & 1) * 16;   // A staging k-offset (0 or 16)
  const int wrow = tid >> 3;         // W staging k-row (0..31)
  const int wcol = (tid & 7) * 16;   // W staging n-offset (0..112)

  for (int kk = 0; kk < K; kk += 32) {
    // ---- stage A tile (128 x 32) as bf16, K-contiguous ----
    {
      uint4 u0, u1;
      if constexpr (A_BF16) {
        const uint4* src = (const uint4*)((const unsigned short*)Av +
                             (size_t)(m0 + srow) * K + kk + sks);
        u0 = src[0]; u1 = src[1];
      } else {
        const float4* src = (const float4*)((const float*)Av +
                             (size_t)(m0 + srow) * K + kk + sks);
        float4 f0 = src[0], f1 = src[1], f2 = src[2], f3 = src[3];
        u0.x = pack2(f0.x, f0.y); u0.y = pack2(f0.z, f0.w);
        u0.z = pack2(f1.x, f1.y); u0.w = pack2(f1.z, f1.w);
        u1.x = pack2(f2.x, f2.y); u1.y = pack2(f2.z, f2.w);
        u1.z = pack2(f3.x, f3.y); u1.w = pack2(f3.z, f3.w);
      }
      uint4* dst = (uint4*)&Alds[srow * AS + sks];
      dst[0] = u0; dst[1] = u1;
    }
    // ---- stage Wt tile (32 x 128) bf16: pure vector copy ----
    {
      const uint4* src = (const uint4*)(WtG + (size_t)(kk + wrow) * N + n0 + wcol);
      uint4* dst = (uint4*)&Wlds[wrow * WS + wcol];
      dst[0] = src[0]; dst[1] = src[1];
    }
    // ---- prefetch next k-step's tiles into cache ----
    if (kk + 32 < K) {
      if constexpr (A_BF16)
        __builtin_prefetch((const unsigned short*)Av +
                           (size_t)(m0 + srow) * K + kk + 32 + sks, 0, 1);
      else
        __builtin_prefetch((const float*)Av +
                           (size_t)(m0 + srow) * K + kk + 32 + sks, 0, 1);
      __builtin_prefetch(WtG + (size_t)(kk + 32 + wrow) * N + n0 + wcol, 0, 1);
    }
    __syncthreads();

    v16bf bfr[2];
#pragma unroll
    for (int nt = 0; nt < 2; ++nt) {
      const __bf16* p = (const __bf16*)&Wlds[wn + nt * 16 + l15];
#pragma unroll
      for (int i = 0; i < 8; ++i) {
        bfr[nt][i]     = p[(sel * 8 + i) * WS];
        bfr[nt][i + 8] = p[(16 + sel * 8 + i) * WS];
      }
    }
#pragma unroll
    for (int ft = 0; ft < 4; ++ft) {
      const unsigned short* ar = &Alds[(wm + ft * 16 + l15) * AS];
      v16bf afr = frag_ld(ar + sel * 8, ar + 16 + sel * 8);
#pragma unroll
      for (int nt = 0; nt < 2; ++nt)
        acc[ft][nt] = __builtin_amdgcn_wmma_f32_16x16x32_bf16(
            false, afr, false, bfr[nt], (short)0, acc[ft][nt], false, false);
    }
    __syncthreads();
  }

  // ---- epilogue: bias, alpha, store ----
#pragma unroll
  for (int ft = 0; ft < 4; ++ft) {
#pragma unroll
    for (int nt = 0; nt < 2; ++nt) {
      const int n  = n0 + wn + nt * 16 + l15;
      const float bn = bias[n];
#pragma unroll
      for (int r = 0; r < 8; ++r) {
        const int m = m0 + wm + ft * 16 + r + 8 * sel;
        const float vv = (acc[ft][nt][r] + bn) * alpha;
        if constexpr (OMODE == 1) {
          const int bb = m / rowsPer, ii = m - bb * rowsPer;
          const int h = n >> 6, d = n & 63;
          ((unsigned short*)Cout)[(((size_t)bb * 16 + h) * rowsPer + ii) * 64 + d]
              = f2bf(vv);
        } else if constexpr (OMODE == 2) {
          const int bb = m / rowsPer, ii = m - bb * rowsPer;
          const int h = n >> 6, d = n & 63;
          ((unsigned short*)Cout)[(((size_t)bb * 16 + h) * 64 + d) * rowsPer + ii]
              = f2bf(vv);
        } else {
          ((float*)Cout)[(size_t)m * N + n] = vv;
        }
      }
    }
  }
}

// ---------------------------------------------------------------------------
// Attention: one block = (b,h) x 128 q-rows.  8 waves, 16 q-rows each.
// K arrives pre-transposed [bh][d=64][key=256], so all staging is vector copy.
// ---------------------------------------------------------------------------
__global__ __launch_bounds__(256) void attn_wmma(
    const unsigned short* __restrict__ Q, const unsigned short* __restrict__ Kt_g,
    const unsigned short* __restrict__ Vb, const int* __restrict__ maskL,
    unsigned short* __restrict__ Ctx)
{
  constexpr int VS = 72;  // padded row stride (ushorts), 144B
  __shared__ __align__(16) unsigned short Vlds[256 * VS];     // 36864 B
  __shared__ float maskAdd[256];                              //  1024 B
  __shared__ __align__(16) unsigned short Scr[8 * 16 * VS];   // 18432 B (Kt / P)

  const int tid = threadIdx.x, wid = tid >> 5, lane = tid & 31;
  const int sel = lane >> 4, l15 = lane & 15;
  const int bh = blockIdx.y, b = bh >> 4, h = bh & 15;
  const int q0 = blockIdx.x * 128;

  const unsigned short* qh = Q    + ((size_t)bh * 4096 + q0) * 64;
  const unsigned short* kh = Kt_g + (size_t)bh * 64 * 256;    // [d][key]
  const unsigned short* vh = Vb   + (size_t)bh * 256 * 64;    // [key][d]

  // ---- stage V (256x64) and the language mask ----
  {
    const uint4* src = (const uint4*)(vh + (size_t)tid * 64);
    uint4* dst = (uint4*)&Vlds[tid * VS];
#pragma unroll
    for (int i = 0; i < 8; ++i) dst[i] = src[i];
    maskAdd[tid] = (maskL[b * 256 + tid] == 0) ? -3.0e38f : 0.0f;
  }

  // ---- q fragments straight from global (row-major, d-contiguous) ----
  v16bf qf[2];
#pragma unroll
  for (int df = 0; df < 2; ++df) {
    const unsigned short* p = qh + (size_t)(wid * 16 + l15) * 64 + df * 32 + sel * 8;
    qf[df] = frag_ld(p, p + 16);
  }

  // ---- S = q·k^T : 16 tiles of 16x16 per wave (full 256 keys) ----
  v8f S[16];
#pragma unroll
  for (int t = 0; t < 16; ++t)
#pragma unroll
    for (int r = 0; r < 8; ++r) S[t][r] = 0.0f;

  unsigned short* Kt = Scr;   // Kt[d][key] for the current 64-key chunk

  for (int c = 0; c < 4; ++c) {
    __syncthreads();   // covers V/mask staging (c==0) and Kt reuse
    {
      const int d = tid >> 2, koff = (tid & 3) * 16;   // vector copy, no transpose
      const uint4* src = (const uint4*)(kh + (size_t)d * 256 + c * 64 + koff);
      uint4* dst = (uint4*)&Kt[d * VS + koff];
      dst[0] = src[0]; dst[1] = src[1];
    }
    __syncthreads();
#pragma unroll
    for (int kt = 0; kt < 4; ++kt) {
      v8f a = S[c * 4 + kt];
#pragma unroll
      for (int df = 0; df < 2; ++df) {
        const __bf16* p = (const __bf16*)&Kt[(df * 32) * VS + kt * 16 + l15];
        v16bf bf;
#pragma unroll
        for (int i = 0; i < 8; ++i) {
          bf[i]     = p[(sel * 8 + i) * VS];
          bf[i + 8] = p[(16 + sel * 8 + i) * VS];
        }
        a = __builtin_amdgcn_wmma_f32_16x16x32_bf16(
                false, qf[df], false, bf, (short)0, a, false, false);
      }
      S[c * 4 + kt] = a;
    }
  }

  __syncthreads();   // all waves done reading Kt before Scr becomes P

  // ---- masked, clamped row softmax; spill P (bf16) to per-wave LDS ----
  float madd[16];
#pragma unroll
  for (int t = 0; t < 16; ++t) madd[t] = maskAdd[t * 16 + l15];

  unsigned short* Pw = Scr + wid * (16 * VS);
  float rinv[8];
#pragma unroll
  for (int r = 0; r < 8; ++r) {
    float mx = -3.4e38f;
#pragma unroll
    for (int t = 0; t < 16; ++t) {
      float s = fminf(fmaxf(S[t][r], -50000.0f), 50000.0f) + madd[t];
      S[t][r] = s;
      mx = fmaxf(mx, s);
    }
#pragma unroll
    for (int off = 1; off < 16; off <<= 1)   // rows live in 16-lane halves
      mx = fmaxf(mx, __shfl_xor(mx, off, 32));
    float sum = 0.0f;
#pragma unroll
    for (int t = 0; t < 16; ++t) {
      const float p = __expf(S[t][r] - mx);
      sum += p;
      Pw[(r + 8 * sel) * VS + t * 16 + l15] = f2bf(p);
    }
#pragma unroll
    for (int off = 1; off < 16; off <<= 1)
      sum += __shfl_xor(sum, off, 32);
    rinv[r] = 1.0f / sum;
  }

  asm volatile("s_wait_dscnt 0" ::: "memory");  // P writes visible to our reads

  // ---- O = P·V ----
  v8f o[4];
#pragma unroll
  for (int dt = 0; dt < 4; ++dt)
#pragma unroll
    for (int r = 0; r < 8; ++r) o[dt][r] = 0.0f;

#pragma unroll
  for (int c = 0; c < 4; ++c) {
#pragma unroll
    for (int kf = 0; kf < 2; ++kf) {
      const unsigned short* ap = Pw + (size_t)l15 * VS + c * 64 + kf * 32;
      v16bf af = frag_ld(ap + sel * 8, ap + 16 + sel * 8);
#pragma unroll
      for (int dt = 0; dt < 4; ++dt) {
        const __bf16* bp = (const __bf16*)&Vlds[(c * 64 + kf * 32) * VS + dt * 16 + l15];
        v16bf bf;
#pragma unroll
        for (int i = 0; i < 8; ++i) {
          bf[i]     = bp[(sel * 8 + i) * VS];
          bf[i + 8] = bp[(16 + sel * 8 + i) * VS];
        }
        o[dt] = __builtin_amdgcn_wmma_f32_16x16x32_bf16(
                    false, af, false, bf, (short)0, o[dt], false, false);
      }
    }
  }

  // ---- normalize and store context bf16 row-major [B][NV][1024] ----
#pragma unroll
  for (int dt = 0; dt < 4; ++dt)
#pragma unroll
    for (int r = 0; r < 8; ++r) {
      const int qrow = q0 + wid * 16 + r + 8 * sel;
      const size_t idx = ((size_t)b * 4096 + qrow) * 1024 + h * 64 + dt * 16 + l15;
      Ctx[idx] = f2bf(o[dt][r] * rinv[r]);
    }
}

// ---------------------------------------------------------------------------
extern "C" void kernel_launch(void* const* d_in, const int* in_sizes, int n_in,
                              void* d_out, int out_size, void* d_ws, size_t ws_size,
                              hipStream_t stream) {
  (void)in_sizes; (void)n_in; (void)out_size; (void)ws_size;
  const float* v     = (const float*)d_in[0];
  const float* l     = (const float*)d_in[1];
  const int*   am    = (const int*)d_in[2];
  const float* w_v   = (const float*)d_in[3];
  const float* b_v   = (const float*)d_in[4];
  const float* w_l   = (const float*)d_in[5];
  const float* b_l   = (const float*)d_in[6];
  const float* w_vl  = (const float*)d_in[7];
  const float* b_vl  = (const float*)d_in[8];
  const float* w_out = (const float*)d_in[9];
  const float* b_out = (const float*)d_in[10];

  char* ws = (char*)d_ws;
  unsigned short* qb   = (unsigned short*)ws; ws += (size_t)32768 * 1024 * 2; // 64 MB
  unsigned short* kb   = (unsigned short*)ws; ws += (size_t)2048  * 1024 * 2; //  4 MB
  unsigned short* vb   = (unsigned short*)ws; ws += (size_t)2048  * 1024 * 2; //  4 MB
  unsigned short* ctx  = (unsigned short*)ws; ws += (size_t)32768 * 1024 * 2; // 64 MB
  unsigned short* wvT  = (unsigned short*)ws; ws += (size_t)1024 * 1024 * 2;  //  2 MB
  unsigned short* wlT  = (unsigned short*)ws; ws += (size_t)768  * 1024 * 2;  // 1.5 MB
  unsigned short* wvlT = (unsigned short*)ws; ws += (size_t)768  * 1024 * 2;  // 1.5 MB
  unsigned short* woT  = (unsigned short*)ws;                                 //  2 MB

  const dim3 blk(256);
  // one-time (per call) weight transpose + bf16 convert
  transpose_w_bf16<<<dim3(16, 16), blk, 0, stream>>>(w_v,   wvT,  1024, 1024);
  transpose_w_bf16<<<dim3(12, 16), blk, 0, stream>>>(w_l,   wlT,  1024, 768);
  transpose_w_bf16<<<dim3(12, 16), blk, 0, stream>>>(w_vl,  wvlT, 1024, 768);
  transpose_w_bf16<<<dim3(16, 16), blk, 0, stream>>>(w_out, woT,  1024, 1024);

  // q = (v @ w_v^T + b_v) * SCALE  -> split bf16 [b][h][4096][64]
  gemm_wmma<0, 1><<<dim3(8, 256), blk, 0, stream>>>(v, wvT, b_v, 0.125f,
                                                    qb, 32768, 1024, 1024, 4096);
  // k = l @ w_l^T + b_l            -> splitT bf16 [b][h][64][256]
  gemm_wmma<0, 2><<<dim3(8, 16),  blk, 0, stream>>>(l, wlT, b_l, 1.0f,
                                                    kb, 2048, 1024, 768, 256);
  // val = l @ w_vl^T + b_vl        -> split bf16 [b][h][256][64]
  gemm_wmma<0, 1><<<dim3(8, 16),  blk, 0, stream>>>(l, wvlT, b_vl, 1.0f,
                                                    vb, 2048, 1024, 768, 256);
  // attention -> ctx bf16 [B][NV][1024]
  attn_wmma<<<dim3(32, 128), blk, 0, stream>>>(qb, kb, vb, am, ctx);
  // out = ctx @ w_out^T + b_out    -> f32 d_out
  gemm_wmma<1, 0><<<dim3(8, 256), blk, 0, stream>>>(ctx, woT, b_out, 1.0f,
                                                    (float*)d_out, 32768, 1024, 1024, 0);
}
